// GraphAttentionLayer_5179730559064
// MI455X (gfx1250) — matrix operand order repaired and checked by reading, not compiled, not run on hardware
//
#include <hip/hip_runtime.h>
#include <hip/hip_bf16.h>

typedef __attribute__((ext_vector_type(16))) _Float16 v16h;
typedef __attribute__((ext_vector_type(8)))  float    v8f;

#define NTOT  4096
#define FIN   128
#define FOUT  64
#define LRALPHA 0.2f
#define LOG2E 1.44269504088896f

// ---------------------------------------------------------------------------
// Kernel 1: Wh = h @ W  (f32 accumulate), store WhT as f16 [b][o][n] (transposed
// so WMMA B-operand loads are contiguous), plus s1 = Wh@a1, s2 = Wh@a2 in f32.
// ---------------------------------------------------------------------------
__global__ __launch_bounds__(64) void gat_wh_kernel(
    const float* __restrict__ h, const float* __restrict__ W,
    const float* __restrict__ a, _Float16* __restrict__ WhT,
    float* __restrict__ s1, float* __restrict__ s2)
{
    __shared__ float hrow[FIN];
    __shared__ float red[4];
    const int tid = threadIdx.x;           // 0..63 == output column o
    const int row = blockIdx.x;            // b*N + n
    const int b   = row >> 12;
    const int n   = row & (NTOT - 1);

    hrow[tid]      = h[(size_t)row * FIN + tid];
    hrow[tid + 64] = h[(size_t)row * FIN + 64 + tid];
    __syncthreads();

    float acc = 0.f;
#pragma unroll 8
    for (int k = 0; k < FIN; ++k)
        acc = fmaf(hrow[k], W[k * FOUT + tid], acc);

    WhT[((size_t)b * FOUT + tid) * NTOT + n] = (_Float16)acc;

    float p1 = acc * a[tid];
    float p2 = acc * a[FOUT + tid];
#pragma unroll
    for (int off = 16; off >= 1; off >>= 1) {
        p1 += __shfl_xor(p1, off, 32);
        p2 += __shfl_xor(p2, off, 32);
    }
    const int wv = tid >> 5;
    if ((tid & 31) == 0) { red[wv] = p1; red[2 + wv] = p2; }
    __syncthreads();
    if (tid == 0) {
        s1[row] = red[0] + red[1];
        s2[row] = red[2] + red[3];
    }
}

// ---------------------------------------------------------------------------
// Kernel 2: fused masked-softmax attention + (P @ Wh) via WMMA f16, + ELU.
// 4 waves / WG; each wave owns a 16-row i-tile. grid = (N/64, B).
// ---------------------------------------------------------------------------
__global__ __launch_bounds__(128) void gat_attn_kernel(
    const int* __restrict__ adj, const _Float16* __restrict__ WhT,
    const float* __restrict__ s1g, const float* __restrict__ s2g,
    float* __restrict__ out)
{
    __shared__ __align__(16) float s2_lds[NTOT];   // 16 KB
    __shared__ float s1_lds[64];
    __shared__ float m_lds[64];
    __shared__ float l_lds[64];

    const int tid    = threadIdx.x;
    const int l      = tid & 31;
    const int wv     = tid >> 5;
    const int b      = blockIdx.y;
    const int ibase0 = blockIdx.x * 64;
    const int ibase  = ibase0 + wv * 16;

    for (int idx = tid; idx < NTOT; idx += 128)
        s2_lds[idx] = s2g[b * NTOT + idx];
    if (tid < 64) s1_lds[tid] = s1g[b * NTOT + ibase0 + tid];
    __syncthreads();

    // ---- pass A: per-row masked max of LeakyReLU(s1[i]+s2[j]) ----
    for (int r = 0; r < 16; ++r) {
        const int   i   = ibase + r;
        const float s1i = s1_lds[wv * 16 + r];
        float mx = -3.0e38f;
        for (int j0 = l * 4; j0 < NTOT; j0 += 128) {
            const int4   a4 = *(const int4*)(adj + (size_t)i * NTOT + j0);
            const float4 s4 = *(const float4*)(&s2_lds[j0]);
            float sc;
            sc = s1i + s4.x; sc = sc >= 0.f ? sc : LRALPHA * sc; if (a4.x > 0) mx = fmaxf(mx, sc);
            sc = s1i + s4.y; sc = sc >= 0.f ? sc : LRALPHA * sc; if (a4.y > 0) mx = fmaxf(mx, sc);
            sc = s1i + s4.z; sc = sc >= 0.f ? sc : LRALPHA * sc; if (a4.z > 0) mx = fmaxf(mx, sc);
            sc = s1i + s4.w; sc = sc >= 0.f ? sc : LRALPHA * sc; if (a4.w > 0) mx = fmaxf(mx, sc);
        }
#pragma unroll
        for (int off = 16; off >= 1; off >>= 1)
            mx = fmaxf(mx, __shfl_xor(mx, off, 32));
        if (l == 0) m_lds[wv * 16 + r] = mx;
    }
    __syncthreads();

    // ---- pass B: exp + WMMA accumulate (flash-attention style) ----
    // A-operand lane mapping (16-bit A 16x32): row = l%16,
// K = hb*16 + {0..15} (elems 0..15)
// -> one contiguous 16-element run: vectorized int4/float4 loads.
    const int   r    = l & 15;
    const int   hb   = l >> 4;
    const int   ocol = l & 15;
    const float s1r  = s1_lds[wv * 16 + r];
    const float nmiL = -m_lds[wv * 16 + r] * LOG2E;   // p = exp2(sc*LOG2E + nmiL)
    const int* const arow = adj + (size_t)(ibase + r) * NTOT + hb * 16;
    const _Float16* const bptr0 =
        WhT + (size_t)b * FOUT * NTOT + (size_t)ocol * NTOT + hb * 16;

    v8f acc0 = {}, acc1 = {}, acc2 = {}, acc3 = {};
    float ls0 = 0.f, ls1 = 0.f, ls2 = 0.f, ls3 = 0.f;

    for (int jbase = 0; jbase < NTOT; jbase += 32) {
        // adj directly from global: each row's 32-int tile span = one 128B line
        const int4 a0 = *(const int4*)(arow + jbase);
        const int4 a1 = *(const int4*)(arow + jbase + 4);
        const int4 a2 = *(const int4*)(arow + jbase + 8);
        const int4 a3 = *(const int4*)(arow + jbase + 12);
        if (jbase + 32 < NTOT)
            __builtin_prefetch(arow + jbase + 32, 0, 3);   // global_prefetch_b8

        const float* const s2p = &s2_lds[jbase + hb * 16];
        const float4 sA = *(const float4*)(s2p);
        const float4 sB = *(const float4*)(s2p + 4);
        const float4 sC = *(const float4*)(s2p + 8);
        const float4 sD = *(const float4*)(s2p + 12);

        // branch-free probability: raw v_exp_f32 + v_cndmask
        float pv[16];
#define GAT_P(dst, aint, s2j, lacc)                                        \
        {                                                                  \
            float sc = s1r + (s2j);                                        \
            sc = sc >= 0.f ? sc : LRALPHA * sc;                            \
            float p = __builtin_amdgcn_exp2f(fmaf(sc, LOG2E, nmiL));       \
            p = ((aint) > 0) ? p : 0.f;                                    \
            (lacc) += p;                                                   \
            (dst) = p;                                                     \
        }
        GAT_P(pv[ 0], a0.x, sA.x, ls0) GAT_P(pv[ 1], a0.y, sA.y, ls1)
        GAT_P(pv[ 2], a0.z, sA.z, ls2) GAT_P(pv[ 3], a0.w, sA.w, ls3)
        GAT_P(pv[ 4], a1.x, sB.x, ls0) GAT_P(pv[ 5], a1.y, sB.y, ls1)
        GAT_P(pv[ 6], a1.z, sB.z, ls2) GAT_P(pv[ 7], a1.w, sB.w, ls3)
        GAT_P(pv[ 8], a2.x, sC.x, ls0) GAT_P(pv[ 9], a2.y, sC.y, ls1)
        GAT_P(pv[10], a2.z, sC.z, ls2) GAT_P(pv[11], a2.w, sC.w, ls3)
        GAT_P(pv[12], a3.x, sD.x, ls0) GAT_P(pv[13], a3.y, sD.y, ls1)
        GAT_P(pv[14], a3.z, sD.z, ls2) GAT_P(pv[15], a3.w, sD.w, ls3)
#undef GAT_P

        v16h av;
#pragma unroll
        for (int e = 0; e < 16; ++e) av[e] = (_Float16)pv[e];

        const _Float16* bp = bptr0 + jbase;
        const v16h b0 = *(const v16h*)(bp);
        const v16h b1 = *(const v16h*)(bp + (size_t)16 * NTOT);
        const v16h b2 = *(const v16h*)(bp + (size_t)32 * NTOT);
        const v16h b3 = *(const v16h*)(bp + (size_t)48 * NTOT);

        acc0 = __builtin_amdgcn_wmma_f32_16x16x32_f16(false, av, false, b0, (short)0, acc0, false, false);
        acc1 = __builtin_amdgcn_wmma_f32_16x16x32_f16(false, av, false, b1, (short)0, acc1, false, false);
        acc2 = __builtin_amdgcn_wmma_f32_16x16x32_f16(false, av, false, b2, (short)0, acc2, false, false);
        acc3 = __builtin_amdgcn_wmma_f32_16x16x32_f16(false, av, false, b3, (short)0, acc3, false, false);
    }

    // ---- row sums: lanes l and l^16 cover the same row ----
    float lsum = (ls0 + ls1) + (ls2 + ls3);
    lsum += __shfl_xor(lsum, 16, 32);
    if (l < 16) l_lds[wv * 16 + l] = lsum;     // same-wave LDS, in-order
    float rden[8];
#pragma unroll
    for (int rr = 0; rr < 8; ++rr) {
        const float s = l_lds[wv * 16 + rr + hb * 8];
        rden[rr] = (s > 0.f) ? (1.f / s) : 0.f;  // guard fully-masked rows
    }

    // ---- finalize: normalize, ELU, store (C/D layout: row rr + 8*hb, col ocol)
    const size_t outbase = ((size_t)b * NTOT + ibase) * FOUT;
#define GAT_ELU(x) ((x) > 0.f ? (x) : (__builtin_amdgcn_exp2f((x) * LOG2E) - 1.f))
#pragma unroll
    for (int rr = 0; rr < 8; ++rr) {
        const size_t rowoff = outbase + (size_t)(rr + hb * 8) * FOUT + ocol;
        float v;
        v = acc0[rr] * rden[rr]; out[rowoff +  0] = GAT_ELU(v);
        v = acc1[rr] * rden[rr]; out[rowoff + 16] = GAT_ELU(v);
        v = acc2[rr] * rden[rr]; out[rowoff + 32] = GAT_ELU(v);
        v = acc3[rr] * rden[rr]; out[rowoff + 48] = GAT_ELU(v);
    }
#undef GAT_ELU
}

// ---------------------------------------------------------------------------
extern "C" void kernel_launch(void* const* d_in, const int* in_sizes, int n_in,
                              void* d_out, int out_size, void* d_ws, size_t ws_size,
                              hipStream_t stream) {
    const float* h   = (const float*)d_in[0];   // (2, 4096, 128) f32
    const int*   adj = (const int*)d_in[1];     // (4096, 4096) i32
    const float* W   = (const float*)d_in[2];   // (128, 64) f32
    const float* a   = (const float*)d_in[3];   // (128, 1) f32
    float* out = (float*)d_out;                 // (2, 4096, 64) f32

    char* ws = (char*)d_ws;
    _Float16* WhT = (_Float16*)ws;                                   // 1 MiB
    float* s1 = (float*)(ws + (size_t)2 * FOUT * NTOT * sizeof(_Float16));
    float* s2 = s1 + 2 * NTOT;                                       // 32 KiB each

    gat_wh_kernel<<<dim3(2 * NTOT), dim3(64), 0, stream>>>(h, W, a, WhT, s1, s2);
    gat_attn_kernel<<<dim3(NTOT / 64, 2), dim3(128), 0, stream>>>(adj, WhT, s1, s2, out);
}
